// AnswerPointerNetwork_58488864637421
// MI455X (gfx1250) — matrix operand order repaired and checked by reading, not compiled, not run on hardware
//
#include <hip/hip_runtime.h>
#include <math.h>

// ---------------------------------------------------------------------------
// AnswerPointerNetwork for MI455X (gfx1250, wave32, WMMA).
//
// Dominant cost: three streaming passes over passEnc (3 x 268 MB ~= 35 us at
// 23.3 TB/s) plus a 34.4 GFLOP GEMM (passEnc @ WPh.T). The GEMM is done with
// v_wmma_f32_16x16x32_bf16 (bf16 inputs, f32 accumulate) fused with the
// tanh/Vt reduction of the pointer step, so passP (134 MB) is never
// materialized and each pointer step reads passEnc exactly once.
// ---------------------------------------------------------------------------

#define HSZ   256     // H
#define D2H   512     // 2H
#define NB    64      // batch
#define LPP   2048    // passage length
#define LQQ   64      // question length

typedef __attribute__((ext_vector_type(16))) __bf16 v16bf;
typedef __attribute__((ext_vector_type(8)))  __bf16 v8bf;
typedef __attribute__((ext_vector_type(8)))  float  v8f;

// ---------------------------------------------------------------------------
// f32 -> bf16 weight conversion (one-time, tiny)
// ---------------------------------------------------------------------------
__global__ void cvt_bf16_kernel(const float* __restrict__ src,
                                __bf16* __restrict__ dst, int n) {
  int i = blockIdx.x * 256 + threadIdx.x;
  if (i < n) dst[i] = (__bf16)src[i];
}

// addQ[b,h] = WQu_b[h] + (VQr @ WQv.T)[h] + WQv_b[h]   (b-invariant, replicated)
__global__ void addq_kernel(const float* __restrict__ VQr,
                            const float* __restrict__ WQvW,
                            const float* __restrict__ WQvb,
                            const float* __restrict__ WQub,
                            float* __restrict__ addQ) {
  int h = threadIdx.x;                       // 256 threads
  const float* w = WQvW + (size_t)h * HSZ;
  float acc = WQvb[h];
  for (int k = 0; k < HSZ; ++k) acc += VQr[k] * w[k];
  float v = acc + WQub[h];
  for (int b = 0; b < NB; ++b) addQ[b * HSZ + h] = v;
}

// add[b,h] = WPh_b[h] + Wah_b[h] + (rQ @ Wah.T)[b,h]
__global__ void addvec_kernel(const float* __restrict__ rQ,
                              const float* __restrict__ WahW,
                              const float* __restrict__ WPhb,
                              const float* __restrict__ Wahb,
                              float* __restrict__ add) {
  int b = blockIdx.x, h = threadIdx.x;       // 64 blocks x 256 threads
  const float* x = rQ + (size_t)b * D2H;
  const float* w = WahW + (size_t)h * D2H;
  float acc = 0.f;
  for (int k = 0; k < D2H; ++k) acc += x[k] * w[k];
  add[b * HSZ + h] = acc + WPhb[h] + Wahb[h];
}

// ---------------------------------------------------------------------------
// Fused WMMA attention-score kernel:
//   s[b, l] = sum_h  tanh( (enc[l,b,:] @ W.T)[h] + add[b,h] ) * vt[b,h]
// enc viewed as [nrows = L*64, 512] row-major; W is bf16 [256, 512].
// Block = 128 threads (4 waves); each wave owns a 16-row M-tile; 64 rows per
// block staged f32->bf16 into a bank-swizzled 64 KB LDS tile.
// ---------------------------------------------------------------------------
__global__ __launch_bounds__(128)
void score_kernel(const float*  __restrict__ enc,
                  const __bf16* __restrict__ Wbf,
                  const float*  __restrict__ add,   // [64,256]
                  const float*  __restrict__ vt,    // [64,256]
                  float*        __restrict__ s_out, // [64, L]
                  int nrows) {
  __shared__ __bf16 tile[64 * 512];                 // exactly 64 KB

  const int t    = threadIdx.x;
  const int wave = t >> 5;
  const int lane = t & 31;
  const int l15  = lane & 15;
  const int hi   = lane >> 4;                       // 0 | 1 (half-wave)
  const long long base = (long long)blockIdx.x * 64;
  const int L = nrows >> 6;                         // rows = l*64 + b

  // ---- stage 64x512 f32 -> bf16 LDS, XOR-swizzled in 16B chunks ----
  for (int i = 0; i < 256; ++i) {
    int idx = t + (i << 7);                         // 0 .. 32767
    int row = idx >> 9;
    int k   = idx & 511;
    int c   = k >> 3;                               // 16B chunk index
    int j   = k & 7;
    int cp  = c ^ (row & 15);                       // bank swizzle
    tile[row * 512 + cp * 8 + j] =
        (__bf16)enc[(base + row) * (long long)D2H + k];
  }
  __syncthreads();

  // A-fragment source row for this lane (ISA 16-bit A layout: lanes 0-15 and
  // 16-31 both map M=0..15; hi selects which K-chunks the lane carries).
  const int arow = wave * 16 + l15;
  const __bf16* aRow = tile + arow * 512;

  float acc_s[8];
#pragma unroll
  for (int r = 0; r < 8; ++r) acc_s[r] = 0.f;

  for (int nt = 0; nt < 16; ++nt) {                 // 16 N-tiles cover H=256
    v8f c = {0.f, 0.f, 0.f, 0.f, 0.f, 0.f, 0.f, 0.f};
    const int ncol = nt * 16 + l15;
    const __bf16* bCol = Wbf + (size_t)ncol * D2H;

#pragma unroll
    for (int ks = 0; ks < 16; ++ks) {               // K = 512 = 16 x 32
      // A frag: two swizzled 16B chunks from LDS
      int c0 = ks * 4 + hi;
      v8bf a0 = *(const v8bf*)(aRow + ((c0       ^ l15) << 3));
      v8bf a1 = *(const v8bf*)(aRow + (((c0 + 2) ^ l15) << 3));
      v16bf a = __builtin_shufflevector(a0, a1, 0, 1, 2, 3, 4, 5, 6, 7,
                                                8, 9, 10, 11, 12, 13, 14, 15);
      // B frag: two 8-K chunks (hi*8, 16+hi*8) of weight row `ncol` (global/L2)
      int kb = ks * 32 + hi * 8;
      v8bf b0 = *(const v8bf*)(bCol + kb);
      v8bf b1 = *(const v8bf*)(bCol + kb + 16);
      v16bf b = __builtin_shufflevector(b0, b1, 0, 1, 2, 3, 4, 5, 6, 7,
                                                8, 9, 10, 11, 12, 13, 14, 15);
      c = __builtin_amdgcn_wmma_f32_16x16x32_bf16(
              false, a, false, b, (short)0, c, false, false);
    }

    // Fused epilogue: tanh(gemm + add[b,n]) * vt[b,n], accumulate over n.
    // C layout: VGPR r -> row (r + 8*hi), column = l15 within this N-tile.
#pragma unroll
    for (int r = 0; r < 8; ++r) {
      int bidx = (int)((base + wave * 16 + r + 8 * hi) & 63);
      float v = tanhf(c[r] + add[bidx * HSZ + ncol]) * vt[bidx * HSZ + ncol];
      acc_s[r] += v;
    }
  }

  // Reduce the 16 column-lanes of each half-wave (h-reduction).
#pragma unroll
  for (int r = 0; r < 8; ++r) {
    float v = acc_s[r];
    v += __shfl_xor(v, 1);
    v += __shfl_xor(v, 2);
    v += __shfl_xor(v, 4);
    v += __shfl_xor(v, 8);
    acc_s[r] = v;
  }
  if (l15 == 0) {                                   // lanes 0 and 16 write
#pragma unroll
    for (int r = 0; r < 8; ++r) {
      long long grow = base + wave * 16 + 8 * hi + r;
      int b = (int)(grow & 63);
      long long l = grow >> 6;
      s_out[(size_t)b * L + l] = acc_s[r];
    }
  }
}

// ---------------------------------------------------------------------------
// question softmax + rQ = a @ quesEnc
// ---------------------------------------------------------------------------
__global__ __launch_bounds__(256)
void qsoftmax_rq_kernel(const float* __restrict__ sQ,
                        const float* __restrict__ quesEnc,
                        float* __restrict__ rQ) {
  int b = blockIdx.x, t = threadIdx.x;
  __shared__ float prob[LQQ];
  float mx = -1e30f;
  for (int q = 0; q < LQQ; ++q) mx = fmaxf(mx, sQ[b * LQQ + q]);
  float sum = 0.f;
  for (int q = 0; q < LQQ; ++q) sum += __expf(sQ[b * LQQ + q] - mx);
  if (t < LQQ) prob[t] = __expf(sQ[b * LQQ + t] - mx) / sum;
  __syncthreads();
  for (int d = t; d < D2H; d += 256) {
    float acc = 0.f;
    for (int q = 0; q < LQQ; ++q)
      acc += prob[q] * quesEnc[((size_t)q * NB + b) * D2H + d];
    rQ[(size_t)b * D2H + d] = acc;
  }
}

// passage softmax -> aP (written straight into d_out slice)
__global__ __launch_bounds__(256)
void psoftmax_kernel(const float* __restrict__ sP, float* __restrict__ aP) {
  int b = blockIdx.x, t = threadIdx.x;
  __shared__ float red[256];
  float mx = -1e30f;
  for (int p = t; p < LPP; p += 256) mx = fmaxf(mx, sP[b * LPP + p]);
  red[t] = mx; __syncthreads();
  for (int s = 128; s > 0; s >>= 1) {
    if (t < s) red[t] = fmaxf(red[t], red[t + s]);
    __syncthreads();
  }
  mx = red[0]; __syncthreads();
  float sum = 0.f;
  for (int p = t; p < LPP; p += 256) sum += __expf(sP[b * LPP + p] - mx);
  red[t] = sum; __syncthreads();
  for (int s = 128; s > 0; s >>= 1) {
    if (t < s) red[t] += red[t + s];
    __syncthreads();
  }
  float inv = 1.f / red[0];
  for (int p = t; p < LPP; p += 256)
    aP[b * LPP + p] = __expf(sP[b * LPP + p] - mx) * inv;
}

// ct partials over p-chunks (deterministic; no float atomics)
__global__ __launch_bounds__(256)
void ct_partial_kernel(const float* __restrict__ aP,
                       const float* __restrict__ passEnc,
                       float* __restrict__ part) {
  int d  = blockIdx.x * 256 + threadIdx.x;          // gridDim.x = 2
  int b  = blockIdx.y;                              // 64
  int pz = blockIdx.z;                              // 8
  float acc = 0.f;
  for (int p = pz * 256; p < pz * 256 + 256; ++p)
    acc += aP[b * LPP + p] * passEnc[((size_t)p * NB + b) * D2H + d];
  part[((size_t)pz * NB + b) * D2H + d] = acc;
}

__global__ void ct_reduce_kernel(const float* __restrict__ part,
                                 float* __restrict__ ct) {
  int i = blockIdx.x * 256 + threadIdx.x;           // 64*512 elements
  float acc = 0.f;
  for (int z = 0; z < 8; ++z) acc += part[(size_t)z * NB * D2H + i];
  ct[i] = acc;
}

// GRUCell(x=rQ, h=ct) -> rQ2    (0.2 GFLOP, VALU)
__global__ __launch_bounds__(256)
void gru_kernel(const float* __restrict__ rQ, const float* __restrict__ ct,
                const float* __restrict__ wih, const float* __restrict__ whh,
                const float* __restrict__ bih, const float* __restrict__ bhh,
                float* __restrict__ rQ2) {
  int b = blockIdx.x, t = threadIdx.x;
  __shared__ float xs[D2H], hs[D2H];
  xs[t] = rQ[(size_t)b * D2H + t];
  xs[t + 256] = rQ[(size_t)b * D2H + t + 256];
  hs[t] = ct[(size_t)b * D2H + t];
  hs[t + 256] = ct[(size_t)b * D2H + t + 256];
  __syncthreads();
  for (int h = t; h < D2H; h += 256) {
    float ir = bih[h], iz = bih[512 + h], in_ = bih[1024 + h];
    float hr = bhh[h], hz = bhh[512 + h], hn = bhh[1024 + h];
    const float* wir = wih + (size_t)h * D2H;
    const float* wiz = wih + (size_t)(512 + h) * D2H;
    const float* win = wih + (size_t)(1024 + h) * D2H;
    const float* whr = whh + (size_t)h * D2H;
    const float* whz = whh + (size_t)(512 + h) * D2H;
    const float* whn = whh + (size_t)(1024 + h) * D2H;
    for (int k = 0; k < D2H; ++k) {
      float x = xs[k], hv = hs[k];
      ir += x * wir[k]; iz += x * wiz[k]; in_ += x * win[k];
      hr += hv * whr[k]; hz += hv * whz[k]; hn += hv * whn[k];
    }
    float r = 1.f / (1.f + __expf(-(ir + hr)));
    float z = 1.f / (1.f + __expf(-(iz + hz)));
    float n = tanhf(in_ + r * hn);
    rQ2[(size_t)b * D2H + h] = (1.f - z) * n + z * hs[h];
  }
}

// ---------------------------------------------------------------------------
extern "C" void kernel_launch(void* const* d_in, const int* in_sizes, int n_in,
                              void* d_out, int out_size, void* d_ws,
                              size_t ws_size, hipStream_t stream) {
  const float* passEnc = (const float*)d_in[0];
  const float* quesEnc = (const float*)d_in[1];
  const float* WQu_W   = (const float*)d_in[2];
  const float* WQu_b   = (const float*)d_in[3];
  const float* WQv_W   = (const float*)d_in[4];
  const float* WQv_b   = (const float*)d_in[5];
  const float* WPh_W   = (const float*)d_in[6];
  const float* WPh_b   = (const float*)d_in[7];
  const float* Wah_W   = (const float*)d_in[8];
  const float* Wah_b   = (const float*)d_in[9];
  const float* Vt1     = (const float*)d_in[10];   // [64,256,1] -> [64,256]
  const float* Vt2     = (const float*)d_in[11];
  const float* VQr     = (const float*)d_in[12];   // [256]
  const float* gru_wih = (const float*)d_in[13];
  const float* gru_whh = (const float*)d_in[14];
  const float* gru_bih = (const float*)d_in[15];
  const float* gru_bhh = (const float*)d_in[16];

  float* aP1 = (float*)d_out;                       // [64,2048]
  float* aP2 = (float*)d_out + (size_t)NB * LPP;

  // workspace carve-up (~2.7 MB total)
  char* w = (char*)d_ws;
  __bf16* WQu_bf = (__bf16*)w;  w += (size_t)HSZ * D2H * 2;
  __bf16* WPh_bf = (__bf16*)w;  w += (size_t)HSZ * D2H * 2;
  float* addQ = (float*)w;      w += (size_t)NB * HSZ * 4;
  float* add1 = (float*)w;      w += (size_t)NB * HSZ * 4;
  float* add2 = (float*)w;      w += (size_t)NB * HSZ * 4;
  float* sQ   = (float*)w;      w += (size_t)NB * LQQ * 4;
  float* sP   = (float*)w;      w += (size_t)NB * LPP * 4;
  float* rQ   = (float*)w;      w += (size_t)NB * D2H * 4;
  float* rQ2  = (float*)w;      w += (size_t)NB * D2H * 4;
  float* ct   = (float*)w;      w += (size_t)NB * D2H * 4;
  float* part = (float*)w;      w += (size_t)8 * NB * D2H * 4;

  const int nW = HSZ * D2H;                          // 131072
  cvt_bf16_kernel<<<(nW + 255) / 256, 256, 0, stream>>>(WQu_W, WQu_bf, nW);
  cvt_bf16_kernel<<<(nW + 255) / 256, 256, 0, stream>>>(WPh_W, WPh_bf, nW);
  addq_kernel<<<1, 256, 0, stream>>>(VQr, WQv_W, WQv_b, WQu_b, addQ);

  // question-aware initial state rQ
  score_kernel<<<(LQQ * NB) / 64, 128, 0, stream>>>(quesEnc, WQu_bf, addQ,
                                                    Vt1, sQ, LQQ * NB);
  qsoftmax_rq_kernel<<<NB, 256, 0, stream>>>(sQ, quesEnc, rQ);

  // pointer step 1
  addvec_kernel<<<NB, 256, 0, stream>>>(rQ, Wah_W, WPh_b, Wah_b, add1);
  score_kernel<<<(LPP * NB) / 64, 128, 0, stream>>>(passEnc, WPh_bf, add1,
                                                    Vt2, sP, LPP * NB);
  psoftmax_kernel<<<NB, 256, 0, stream>>>(sP, aP1);
  ct_partial_kernel<<<dim3(2, NB, 8), 256, 0, stream>>>(aP1, passEnc, part);
  ct_reduce_kernel<<<(NB * D2H) / 256, 256, 0, stream>>>(part, ct);

  // GRU update + pointer step 2
  gru_kernel<<<NB, 256, 0, stream>>>(rQ, ct, gru_wih, gru_whh, gru_bih,
                                     gru_bhh, rQ2);
  addvec_kernel<<<NB, 256, 0, stream>>>(rQ2, Wah_W, WPh_b, Wah_b, add2);
  score_kernel<<<(LPP * NB) / 64, 128, 0, stream>>>(passEnc, WPh_bf, add2,
                                                    Vt2, sP, LPP * NB);
  psoftmax_kernel<<<NB, 256, 0, stream>>>(sP, aP2);
}
